// DWT_WindowAttention_SW_17042430230761
// MI455X (gfx1250) — compile-verified, matrix operand order checked by reading, not run on hardware
//
#include <hip/hip_runtime.h>
#include <hip/hip_bf16.h>

typedef __attribute__((ext_vector_type(16))) _Float16 v16h;
typedef __attribute__((ext_vector_type(8)))  float    v8f;

// ---------------------------------------------------------------------------
// WMMA helpers (v_wmma_f32_16x16x32_f16, wave32)
// Fragment layout (CDNA5 ISA 7.12.2, 16-bit A 16x32):
//   lane l: row = l&15, half = l>>4
//   VGPR j (j<4): k = 2j + 8*half ; VGPR j (j>=4): k = 16 + 2(j-4) + 8*half
// B mirrors A with N in place of M. LDS tiles stored [row][k], padded stride
// 36 halves -> conflict-free b32 pair loads.
// ---------------------------------------------------------------------------
__device__ __forceinline__ v8f zero_v8f() {
  v8f z;
#pragma unroll
  for (int i = 0; i < 8; ++i) z[i] = 0.0f;
  return z;
}

__device__ __forceinline__ v8f wmma_f16(v16h a, v16h b, v8f c) {
  return __builtin_amdgcn_wmma_f32_16x16x32_f16(false, a, false, b, (short)0, c,
                                                false, false);
}

__device__ __forceinline__ v16h ld_frag(const _Float16* __restrict__ base,
                                        int row0, int stride, int kbase) {
  const int lane = threadIdx.x & 31;
  const int row  = row0 + (lane & 15);
  const int half = (lane >> 4) << 3;     // 0 or 8
  const _Float16* p = base + row * stride + kbase + half;
  v16h f;
#pragma unroll
  for (int j = 0; j < 4; ++j) { f[2*j]   = p[2*j];      f[2*j+1]   = p[2*j+1]; }
#pragma unroll
  for (int j = 0; j < 4; ++j) { f[8+2*j] = p[16+2*j];   f[8+2*j+1] = p[16+2*j+1]; }
  return f;
}

// K=16 fragment (upper 16 K-slots zero) for the cosine-similarity GEMM
__device__ __forceinline__ v16h ld_frag_k16(const _Float16* __restrict__ base,
                                            int row0, int stride) {
  const int lane = threadIdx.x & 31;
  const int row  = row0 + (lane & 15);
  const int half = (lane >> 4) << 3;
  const _Float16* p = base + row * stride + half;
  v16h f;
#pragma unroll
  for (int i = 0; i < 16; ++i) f[i] = (_Float16)0.0f;
#pragma unroll
  for (int j = 0; j < 4; ++j) { f[2*j] = p[2*j]; f[2*j+1] = p[2*j+1]; }
  return f;
}

// ---------------------------------------------------------------------------
// Async global->LDS copy (CDNA5 Ch.8, tracked by ASYNCcnt).
// vdst VGPR carries the logical LDS byte address (generic-pointer low 32 bits
// per the LDS aperture rule); saddr form: mem = SGPR64 + VGPR32 offset.
// ---------------------------------------------------------------------------
__device__ __forceinline__ void async_g2l_b128(uint32_t lds_byte_addr,
                                               const void* gbase,
                                               uint32_t gbyte_off) {
  asm volatile("global_load_async_to_lds_b128 %0, %1, %2"
               :
               : "v"(lds_byte_addr), "v"(gbyte_off), "s"(gbase)
               : "memory");
}
__device__ __forceinline__ void wait_asynccnt0() {
  asm volatile("s_wait_asynccnt 0x0" ::: "memory");
}

// ---------------------------------------------------------------------------
// Constants
// ---------------------------------------------------------------------------
#define BATCH   8
#define DIM     64
#define H2      128
#define W2      128
#define PLANE   16384          // 128*128
#define NPIX    (BATCH*DIM*PLANE)
#define ASTRIDE 36             // padded halves per m-row in packed A tiles
#define ACHUNK  (64*ASTRIDE)   // halves per K-chunk tile (2304 -> 4608 B)

// ---------------------------------------------------------------------------
// Weight packing: f32 OIHW -> f16 K-chunk tiles [chunk][m(64)][36]
// (pad halves zeroed so the whole chunk can be async-copied verbatim to LDS)
// ---------------------------------------------------------------------------
__global__ void pack3x3_kernel(const float* __restrict__ w,
                               _Float16* __restrict__ dst, int groups) {
  int idx = blockIdx.x * 256 + threadIdx.x;
  int total = groups * 18 * ACHUNK;
  if (idx >= total) return;
  int kk = idx % ASTRIDE;
  int m  = (idx / ASTRIDE) & 63;
  int cg = idx / ACHUNK;
  int ch = cg % 18, g = cg / 18;
  int tap = ch >> 1, icb = (ch & 1) << 5;
  _Float16 v = (_Float16)0.0f;
  if (kk < 32) v = (_Float16)w[((g * 64 + m) * 64 + icb + kk) * 9 + tap];
  dst[idx] = v;
}

__global__ void pack1x1_kernel(const float* __restrict__ w,
                               _Float16* __restrict__ dst, int cin, int moff,
                               int chunks) {
  int idx = blockIdx.x * 256 + threadIdx.x;
  int total = chunks * ACHUNK;
  if (idx >= total) return;
  int kk = idx % ASTRIDE;
  int m  = (idx / ASTRIDE) & 63;
  int ch = idx / ACHUNK;
  _Float16 v = (_Float16)0.0f;
  if (kk < 32) v = (_Float16)w[(moff + m) * cin + ch * 32 + kk];
  dst[idx] = v;
}

// ---------------------------------------------------------------------------
// Haar DWT: x(8,64,256,256) f32 -> ll f32+f16, lh/hl/hh f16 (each 8,64,128,128)
// ---------------------------------------------------------------------------
__global__ void dwt_kernel(const float* __restrict__ x, float* __restrict__ ll32,
                           _Float16* __restrict__ ll16, _Float16* __restrict__ lh,
                           _Float16* __restrict__ hl, _Float16* __restrict__ hh) {
  int idx = blockIdx.x * 256 + threadIdx.x;
  if (idx >= NPIX) return;
  int j = idx & 127, i = (idx >> 7) & 127;
  int bc = idx >> 14;
  const float* p = x + (size_t)bc * 65536 + i * 512 + j * 2;
  float2 t = *(const float2*)p;
  float2 u = *(const float2*)(p + 256);
  float a = t.x, b = t.y, c = u.x, d = u.y;
  float llv = (a + b + c + d) * 0.5f;
  float lhv = (a - b + c - d) * 0.5f;
  float hlv = (a + b - c - d) * 0.5f;
  float hhv = (a - b - c + d) * 0.5f;
  ll32[idx] = llv;
  ll16[idx] = (_Float16)llv;
  lh[idx] = (_Float16)lhv;
  hl[idx] = (_Float16)hlv;
  hh[idx] = (_Float16)hhv;
}

// ---------------------------------------------------------------------------
// Grouped 3x3 conv as implicit GEMM. One block = one (b, g, image row y).
// M=64 out-ch, N=128 (full row), K=576 in 18 chunks of 32. 8 waves; A tile is
// async-copied global->LDS, B tile is an im2col transpose-scatter.
// ---------------------------------------------------------------------------
template <typename OutT, bool RELU, int GROUPS>
__global__ __launch_bounds__(256) void conv3x3_wmma(
    const _Float16* __restrict__ in0, const _Float16* __restrict__ in1,
    const _Float16* __restrict__ in2, const _Float16* __restrict__ wpack,
    OutT* __restrict__ out) {
  __shared__ __attribute__((aligned(16))) _Float16 aT[ACHUNK];
  __shared__ __attribute__((aligned(16))) _Float16 bT[128 * 36];
  int bid = blockIdx.x;
  int y = bid & 127;
  int g = (bid >> 7) % GROUPS;
  int b = bid / (128 * GROUPS);
  const _Float16* src = (g == 0) ? in0 : ((g == 1) ? in1 : in2);
  src += (size_t)b * DIM * PLANE;
  const _Float16* wg = wpack + (size_t)g * 18 * ACHUNK;

  int tid = threadIdx.x;
  int wave = tid >> 5, lane = tid & 31;
  int n0 = wave << 4;
  uint32_t aT_lds = (uint32_t)(uintptr_t)aT;
  v8f acc[4];
#pragma unroll
  for (int mi = 0; mi < 4; ++mi) acc[mi] = zero_v8f();

  for (int ch = 0; ch < 18; ++ch) {
    int tap = ch >> 1, icb = (ch & 1) << 5;
    int dy = tap / 3 - 1, dx = tap % 3 - 1;
    {  // stage A tile: async copy of 4608 contiguous bytes (pre-padded pack)
      const void* ws = (const void*)(wg + (size_t)ch * ACHUNK);
      uint32_t o0 = (uint32_t)tid << 4;          // 256 lanes * 16B = 4096
      async_g2l_b128(aT_lds + o0, ws, o0);
      if (tid < 32) {
        uint32_t o1 = 4096u + ((uint32_t)tid << 4);  // tail 512B
        async_g2l_b128(aT_lds + o1, ws, o1);
      }
    }
    {  // stage B tile: im2col of one shifted image row, transposed to [n][k]
      int kk = tid >> 3;
      int seg = (tid & 7) << 4;
      int yy = y + dy;
      bool yok = (yy >= 0) && (yy < H2);
      const _Float16* row = src + (size_t)(icb + kk) * PLANE + yy * W2;
      __builtin_prefetch(row + PLANE, 0, 1);  // next channel (kk+1), same row
#pragma unroll
      for (int e = 0; e < 16; ++e) {
        int xs = seg + e + dx;
        _Float16 val = (_Float16)0.0f;
        if (yok && xs >= 0 && xs < W2) val = row[xs];
        bT[(seg + e) * 36 + kk] = val;
      }
    }
    wait_asynccnt0();
    __syncthreads();
    v16h bf = ld_frag(bT, n0, 36, 0);
#pragma unroll
    for (int mi = 0; mi < 4; ++mi) {
      v16h af = ld_frag(aT, mi << 4, ASTRIDE, 0);
      acc[mi] = wmma_f16(af, bf, acc[mi]);
    }
    __syncthreads();
  }
  int lm = lane & 15, hs = (lane >> 4) << 3;
#pragma unroll
  for (int mi = 0; mi < 4; ++mi) {
#pragma unroll
    for (int r = 0; r < 8; ++r) {
      int m = (mi << 4) + r + hs;
      float v = acc[mi][r];
      if (RELU) v = fmaxf(v, 0.0f);
      out[((size_t)(b * (DIM * GROUPS) + g * DIM + m) << 14) + (y << 7) + n0 + lm] =
          (OutT)v;
    }
  }
}

// ---------------------------------------------------------------------------
// 1x1 conv as GEMM. Same tiling; A tile async-copied, B tile vector copy.
// ---------------------------------------------------------------------------
template <int CHUNKS, typename OutT, bool RELU>
__global__ __launch_bounds__(256) void conv1x1_wmma(
    const _Float16* __restrict__ in, const _Float16* __restrict__ wpack,
    OutT* __restrict__ out) {
  constexpr int CIN = CHUNKS * 32;
  __shared__ __attribute__((aligned(16))) _Float16 aT[ACHUNK];
  __shared__ __attribute__((aligned(16))) _Float16 bT[128 * 36];
  int y = blockIdx.x & 127, b = blockIdx.x >> 7;
  int tid = threadIdx.x;
  int wave = tid >> 5, lane = tid & 31;
  int n0 = wave << 4;
  uint32_t aT_lds = (uint32_t)(uintptr_t)aT;
  v8f acc[4];
#pragma unroll
  for (int mi = 0; mi < 4; ++mi) acc[mi] = zero_v8f();

  for (int ch = 0; ch < CHUNKS; ++ch) {
    {  // async A stage
      const void* ws = (const void*)(wpack + (size_t)ch * ACHUNK);
      uint32_t o0 = (uint32_t)tid << 4;
      async_g2l_b128(aT_lds + o0, ws, o0);
      if (tid < 32) {
        uint32_t o1 = 4096u + ((uint32_t)tid << 4);
        async_g2l_b128(aT_lds + o1, ws, o1);
      }
    }
    {
      int kk = tid >> 3;
      int seg = (tid & 7) << 4;
      const _Float16* row =
          in + ((size_t)(b * CIN + ch * 32 + kk) << 14) + (y << 7) + seg;
      __builtin_prefetch(row + (PLANE * 32), 0, 1);
      uint4 dv = *(const uint4*)row;
      const _Float16* hv = (const _Float16*)&dv;
#pragma unroll
      for (int e = 0; e < 16; ++e) bT[(seg + e) * 36 + kk] = hv[e];
    }
    wait_asynccnt0();
    __syncthreads();
    v16h bf = ld_frag(bT, n0, 36, 0);
#pragma unroll
    for (int mi = 0; mi < 4; ++mi) {
      v16h af = ld_frag(aT, mi << 4, ASTRIDE, 0);
      acc[mi] = wmma_f16(af, bf, acc[mi]);
    }
    __syncthreads();
  }
  int lm = lane & 15, hs = (lane >> 4) << 3;
#pragma unroll
  for (int mi = 0; mi < 4; ++mi) {
#pragma unroll
    for (int r = 0; r < 8; ++r) {
      int m = (mi << 4) + r + hs;
      float v = acc[mi][r];
      if (RELU) v = fmaxf(v, 0.0f);
      out[((size_t)(b * DIM + m) << 14) + (y << 7) + n0 + lm] = (OutT)v;
    }
  }
}

// ---------------------------------------------------------------------------
// Depthwise 3x3 on u (qkv V-part) fused with v = dw*(1+filt)
// ---------------------------------------------------------------------------
__global__ void dwconv_fuse_kernel(const _Float16* __restrict__ u,
                                   const _Float16* __restrict__ filt,
                                   const float* __restrict__ wdw,
                                   _Float16* __restrict__ v) {
  int idx = blockIdx.x * 256 + threadIdx.x;
  if (idx >= NPIX) return;
  int x = idx & 127, y = (idx >> 7) & 127;
  int c = (idx >> 14) & 63;
  const float* wp = wdw + (size_t)(128 + c) * 9;
  const _Float16* up = u + ((size_t)(idx >> 14) << 14);
  float acc = 0.0f;
#pragma unroll
  for (int ky = 0; ky < 3; ++ky) {
    int yy = y + ky - 1;
    if (yy < 0 || yy >= H2) continue;
#pragma unroll
    for (int kx = 0; kx < 3; ++kx) {
      int xx = x + kx - 1;
      if (xx < 0 || xx >= W2) continue;
      acc += wp[ky * 3 + kx] * (float)up[yy * W2 + xx];
    }
  }
  float f = (float)filt[idx];
  v[idx] = (_Float16)(acc * f + acc);
}

// ---------------------------------------------------------------------------
// Window attention: one wave (32 threads) per (window, head).
//   scores = qn^T qn (K=16 via zero-padded 16x16x32 WMMA) + bias, *temp,
//   softmax rows in LDS (in place), out = V (16x64) @ P^T via 2 K-chunks.
// Output written with window-reverse + roll(+4,+4), f16.
// ---------------------------------------------------------------------------
__global__ __launch_bounds__(32) void attn_kernel(
    const float* __restrict__ ll, const _Float16* __restrict__ vbuf,
    const float* __restrict__ relb, const float* __restrict__ temperature,
    _Float16* __restrict__ out) {
  __shared__ float    qs[16 * 65];    // [d][n] f32 (rolled q window)
  __shared__ _Float16 qt[64 * 18];    // [n][d] normalized f16
  __shared__ _Float16 vs[16 * 66];    // [d][m] f16
  __shared__ _Float16 sst[64 * 66];   // [n][m] scores -> probs (in place)

  int bx = blockIdx.x;
  int h = bx & 3;
  int wid = bx >> 2;
  int wx = wid & 15, wy = (wid >> 4) & 15, b = wid >> 8;
  int tid = threadIdx.x;

  // stage q (with -SHIFT roll) and v windows
  for (int i = tid; i < 1024; i += 32) {
    int d = i >> 6, n = i & 63;
    int iy = n >> 3, ix = n & 7;
    size_t plane = (size_t)(b * DIM + h * 16 + d) << 14;
    int yq = ((wy << 3) + iy + 4) & 127;
    int xq = ((wx << 3) + ix + 4) & 127;
    qs[d * 65 + n] = ll[plane + (yq << 7) + xq];
    int yv = (wy << 3) + iy, xv = (wx << 3) + ix;
    vs[d * 66 + n] = vbuf[plane + (yv << 7) + xv];
  }
  __syncthreads();

  // per-column L2 normalization of q (over d=16)
#pragma unroll
  for (int rr = 0; rr < 2; ++rr) {
    int n = tid + rr * 32;
    float s = 0.0f;
#pragma unroll
    for (int d = 0; d < 16; ++d) {
      float qv = qs[d * 65 + n];
      s += qv * qv;
    }
    float inv = 1.0f / fmaxf(sqrtf(s), 1e-12f);
#pragma unroll
    for (int d = 0; d < 16; ++d)
      qt[n * 18 + d] = (_Float16)(qs[d * 65 + n] * inv);
  }
  __syncthreads();

  // scores = qn^T qn  (M=n, N=m, K=16)
  v16h qf[4];
#pragma unroll
  for (int i = 0; i < 4; ++i) qf[i] = ld_frag_k16(qt, i << 4, 18);
  v8f sc[4][4];
#pragma unroll
  for (int tn = 0; tn < 4; ++tn)
#pragma unroll
    for (int tm = 0; tm < 4; ++tm)
      sc[tn][tm] = wmma_f16(qf[tn], qf[tm], zero_v8f());

  int lm = tid & 15, hs = (tid >> 4) << 3;
  float tscale = temperature[h];
#pragma unroll
  for (int tn = 0; tn < 4; ++tn)
#pragma unroll
    for (int tm = 0; tm < 4; ++tm)
#pragma unroll
      for (int r = 0; r < 8; ++r) {
        int n = (tn << 4) + r + hs;
        int m = (tm << 4) + lm;
        int rid = ((n >> 3) - (m >> 3) + 7) * 15 + ((n & 7) - (m & 7) + 7);
        float bias = relb[rid * 4 + h];
        sst[n * 66 + m] = (_Float16)((sc[tn][tm][r] + bias) * tscale);
      }
  __syncthreads();

  // softmax over m, in place
#pragma unroll
  for (int rr = 0; rr < 2; ++rr) {
    int n = tid + rr * 32;
    _Float16* row = &sst[n * 66];
    float mx = -3.0e38f;
    for (int m = 0; m < 64; ++m) mx = fmaxf(mx, (float)row[m]);
    float sum = 0.0f;
    for (int m = 0; m < 64; ++m) {
      float e = __expf((float)row[m] - mx);
      sum += e;
      row[m] = (_Float16)e;
    }
    float is = 1.0f / sum;
    for (int m = 0; m < 64; ++m) row[m] = (_Float16)((float)row[m] * is);
  }
  __syncthreads();

  // out[d][n] = sum_m V[d][m] P[n][m]  (M=d, N=n, K=m=64 -> 2 chunks)
  v8f acc2[4];
#pragma unroll
  for (int tn = 0; tn < 4; ++tn) acc2[tn] = zero_v8f();
#pragma unroll
  for (int kc = 0; kc < 2; ++kc) {
    int k0 = kc << 5;
    v16h av = ld_frag(vs, 0, 66, k0);
#pragma unroll
    for (int tn = 0; tn < 4; ++tn) {
      v16h bf = ld_frag(sst, tn << 4, 66, k0);
      acc2[tn] = wmma_f16(av, bf, acc2[tn]);
    }
  }

  // store with window reverse + roll(+4,+4)
#pragma unroll
  for (int tn = 0; tn < 4; ++tn)
#pragma unroll
    for (int r = 0; r < 8; ++r) {
      int d = r + hs;
      int n = (tn << 4) + lm;
      int iy = n >> 3, ix = n & 7;
      int yo = ((wy << 3) + iy + 4) & 127;
      int xo = ((wx << 3) + ix + 4) & 127;
      out[((size_t)(b * DIM + h * 16 + d) << 14) + (yo << 7) + xo] =
          (_Float16)acc2[tn][r];
    }
}

// ---------------------------------------------------------------------------
// Inverse Haar DWT: (proj, yh[lh2|hl2|hh2]) -> output (8,64,256,256) f32
// ---------------------------------------------------------------------------
__global__ void idwt_kernel(const float* __restrict__ llp,
                            const float* __restrict__ yh,
                            float* __restrict__ out) {
  int idx = blockIdx.x * 256 + threadIdx.x;
  if (idx >= NPIX) return;
  int j = idx & 127, i = (idx >> 7) & 127;
  int c = (idx >> 14) & 63, b = idx >> 20;
  float ll = llp[idx];
  size_t base = ((size_t)(b * 192 + c) << 14) + (i << 7) + j;
  float lh = yh[base];
  float hl = yh[base + ((size_t)64 << 14)];
  float hh = yh[base + ((size_t)128 << 14)];
  float a  = (ll + lh + hl + hh) * 0.5f;
  float bb = (ll - lh + hl - hh) * 0.5f;
  float cc = (ll + lh - hl - hh) * 0.5f;
  float dd = (ll - lh - hl + hh) * 0.5f;
  float* o = out + ((size_t)(b * DIM + c) << 16) + i * 512 + j * 2;
  *(float2*)o         = make_float2(a, bb);
  *(float2*)(o + 256) = make_float2(cc, dd);
}

// ---------------------------------------------------------------------------
// Launcher
// ---------------------------------------------------------------------------
extern "C" void kernel_launch(void* const* d_in, const int* in_sizes, int n_in,
                              void* d_out, int out_size, void* d_ws,
                              size_t ws_size, hipStream_t stream) {
  const float* x         = (const float*)d_in[0];
  const float* temp      = (const float*)d_in[1];
  const float* relb      = (const float*)d_in[2];
  const float* w_high1   = (const float*)d_in[3];
  const float* w_high2   = (const float*)d_in[4];
  const float* w_highout = (const float*)d_in[5];
  const float* w_qkv     = (const float*)d_in[6];
  const float* w_dwconv  = (const float*)d_in[7];
  const float* w_proj    = (const float*)d_in[8];

  char* wp = (char*)d_ws;
  auto carve = [&](size_t bytes) -> void* {
    void* p = (void*)wp;
    wp += (bytes + 255) & ~(size_t)255;
    return p;
  };
  const size_t Q = (size_t)NPIX;  // 8*64*128*128
  float*    ll32   = (float*)carve(Q * 4);
  _Float16* ll16   = (_Float16*)carve(Q * 2);
  _Float16* lh16   = (_Float16*)carve(Q * 2);
  _Float16* hl16   = (_Float16*)carve(Q * 2);
  _Float16* hh16   = (_Float16*)carve(Q * 2);
  _Float16* t1     = (_Float16*)carve(Q * 2 * 2);        // 128 channels
  _Float16* filt16 = (_Float16*)carve(Q * 2);
  _Float16* u16    = (_Float16*)carve(Q * 2);
  _Float16* v16buf = (_Float16*)carve(Q * 2);
  _Float16* ao16   = (_Float16*)carve(Q * 2);
  float*    proj32 = (float*)carve(Q * 4);
  float*    yh32   = (float*)carve(Q * 4 * 3);           // 192 channels
  _Float16* packH1 = (_Float16*)carve((size_t)2 * 18 * ACHUNK * 2);
  _Float16* packHO = (_Float16*)carve((size_t)3 * 18 * ACHUNK * 2);
  _Float16* packH2 = (_Float16*)carve((size_t)4 * ACHUNK * 2);
  _Float16* packQ  = (_Float16*)carve((size_t)2 * ACHUNK * 2);
  _Float16* packP  = (_Float16*)carve((size_t)2 * ACHUNK * 2);

  // weight packing (f32 -> f16 padded K-chunk tiles)
  pack3x3_kernel<<<(2 * 18 * ACHUNK + 255) / 256, 256, 0, stream>>>(w_high1, packH1, 2);
  pack3x3_kernel<<<(3 * 18 * ACHUNK + 255) / 256, 256, 0, stream>>>(w_highout, packHO, 3);
  pack1x1_kernel<<<(4 * ACHUNK + 255) / 256, 256, 0, stream>>>(w_high2, packH2, 128, 0, 4);
  pack1x1_kernel<<<(2 * ACHUNK + 255) / 256, 256, 0, stream>>>(w_qkv, packQ, 64, 128, 2);
  pack1x1_kernel<<<(2 * ACHUNK + 255) / 256, 256, 0, stream>>>(w_proj, packP, 64, 0, 2);

  // DWT
  dwt_kernel<<<NPIX / 256, 256, 0, stream>>>(x, ll32, ll16, lh16, hl16, hh16);

  // high-frequency filter branch: t1 = relu(g3x3(lh|hl)); filt = relu(1x1(t1))
  conv3x3_wmma<_Float16, true, 2><<<8 * 2 * 128, 256, 0, stream>>>(
      lh16, hl16, nullptr, packH1, t1);
  conv1x1_wmma<4, _Float16, true><<<8 * 128, 256, 0, stream>>>(t1, packH2, filt16);

  // qkv V-part only (q/k of qkv are dead in the reference graph)
  conv1x1_wmma<2, _Float16, false><<<8 * 128, 256, 0, stream>>>(ll16, packQ, u16);
  dwconv_fuse_kernel<<<NPIX / 256, 256, 0, stream>>>(u16, filt16, w_dwconv, v16buf);

  // shifted-window cosine attention
  attn_kernel<<<2048 * 4, 32, 0, stream>>>(ll32, v16buf, relb, temp, ao16);

  // output projection
  conv1x1_wmma<2, float, false><<<8 * 128, 256, 0, stream>>>(ao16, packP, proj32);

  // high-frequency output branch: yh = relu(g3x3(lh|hl|hh))
  conv3x3_wmma<float, true, 3><<<8 * 3 * 128, 256, 0, stream>>>(
      lh16, hl16, hh16, packHO, yh32);

  // IDWT -> final output
  idwt_kernel<<<NPIX / 256, 256, 0, stream>>>(proj32, yh32, (float*)d_out);

  (void)in_sizes; (void)n_in; (void)out_size; (void)ws_size;
}
